// TorchHSTUAttention_46755013984916
// MI455X (gfx1250) — matrix operand order repaired and checked by reading, not compiled, not run on hardware
//
#include <hip/hip_runtime.h>
#include <hip/hip_bf16.h>

typedef __attribute__((ext_vector_type(16))) __bf16 v16bf;
typedef __attribute__((ext_vector_type(8)))  __bf16 v8bf;
typedef __attribute__((ext_vector_type(8)))  float  v8f;

#define HH  8
#define DD  64
#define DVV 64
#define WAVES 4            // 4 waves/block
#define QPW  32            // 2 x 16-query tiles per wave
#define QBLK (WAVES*QPW)   // 128 queries per block
#define QBLKS 8            // MAX_SEQLEN(1024) / 128

__device__ __forceinline__ v16bf cat8(v8bf lo, v8bf hi) {
  return __builtin_shufflevector(lo, hi, 0,1,2,3,4,5,6,7,8,9,10,11,12,13,14,15);
}

__global__ __launch_bounds__(128) void hstu_attn_kernel(
    const float* __restrict__ tq, const float* __restrict__ tk,
    const float* __restrict__ tv, const int* __restrict__ offsets,
    const int* __restrict__ max_seqlen_p, const int* __restrict__ num_cand,
    const int* __restrict__ num_ctx, float* __restrict__ out)
{
  const int tid  = threadIdx.x;
  const int wave = tid >> 5;
  const int lane = tid & 31;

  const int qb = blockIdx.x & (QBLKS - 1);
  const int h  = (blockIdx.x >> 3) & (HH - 1);
  const int b  = blockIdx.x >> 6;

  const int off = offsets[b];
  const int len = offsets[b + 1] - off;
  if (qb * QBLK >= len) return;                 // uniform over block

  const int nctx    = num_ctx[b];
  const int max_ids = len - nctx + 1 - num_cand[b];
  const float rN    = 1.0f / (float)max_seqlen_p[0];

  const int q0 = qb * QBLK + wave * QPW;        // this wave's 32-query range

  // causal bound: rows only reach cols <= row (eye/dist/candidate rules),
  // except adj==0 rows (< nctx) which reach ~everything.
  const int kend = (qb * QBLK < nctx) ? len : min(len, qb * QBLK + QBLK);

  const int m  = lane & 15;                     // A row / B,C column
  const int kh = lane >> 4;                     // lane half

  // block-cooperative staging coords: 128 threads = 8 rows x 16 float4/row
  const int srow  = tid >> 4;                   // 0..7
  const int scol4 = (tid & 15) << 2;            // 0..60

  // uniform 64-bit bases; per-row offsets stay 32-bit
  const float* __restrict__ qbase = tq + (size_t)off * (HH * DD)  + h * DD;
  const float* __restrict__ kbase = tk + (size_t)off * (HH * DD)  + h * DD;
  const float* __restrict__ vbase = tv + (size_t)off * (HH * DVV) + h * DVV;
  float*       __restrict__ obase = out + (size_t)off * (HH * DVV) + h * DVV;

  __shared__ __align__(32) __bf16 Qs[QBLK][DD];      // 16 KB
  __shared__ __align__(32) __bf16 Ks[32][DD];        //  4 KB
  __shared__ __align__(32) __bf16 Vt[DVV][32];       //  4 KB, [dv][key]
  __shared__ __align__(32) float  Ss[WAVES][32][32]; // 16 KB, per-wave 2 tiles

  // ---- stage 128-query Q block (clamped rows; OOB rows masked at store) ----
  #pragma unroll
  for (int r = 0; r < QBLK; r += 8) {
    const int row = r + srow;
    const unsigned gq = (unsigned)min(qb * QBLK + row, len - 1);
    const float4 v = *(const float4*)(qbase + (gq * (unsigned)(HH * DD) + scol4));
    Qs[row][scol4 + 0] = (__bf16)v.x;
    Qs[row][scol4 + 1] = (__bf16)v.y;
    Qs[row][scol4 + 2] = (__bf16)v.z;
    Qs[row][scol4 + 3] = (__bf16)v.w;
  }
  __syncthreads();

  // ---- per-wave Q A-fragments for both tiles (contiguous k = 16*kh + s order) ----
  v16bf aq[2][2];
  bool  tact[2], trz[2];
  int   adjr[2][8];
  bool  rvalid[2][8], rzero[2][8];
  #pragma unroll
  for (int t = 0; t < 2; ++t) {
    const int q0t = q0 + 16 * t;
    tact[t] = (q0t < len);
    trz[t]  = (q0t < nctx);
    const __bf16* qrow = &Qs[wave * QPW + 16 * t + m][0];
    aq[t][0] = cat8(*(const v8bf*)(qrow +       kh * 16), *(const v8bf*)(qrow + 8 + kh * 16));
    aq[t][1] = cat8(*(const v8bf*)(qrow + 32 +  kh * 16), *(const v8bf*)(qrow + 40 + kh * 16));
    #pragma unroll
    for (int j = 0; j < 8; ++j) {
      const int rowq = q0t + j + 8 * kh;
      int a = rowq - nctx + 1;
      a = a > 0 ? a : 0;
      a = a < max_ids ? a : max_ids;
      adjr[t][j]   = a;
      rzero[t][j]  = (a == 0);
      rvalid[t][j] = (rowq < len);
    }
  }

  v8f acc[2][4] = {};

  for (int k0 = 0; k0 < kend; k0 += 32) {
    __syncthreads();   // previous chunk's Ks/Vt reads complete

    // ---- block-cooperative stage of 32-key K + transposed V (clamped) ----
    #pragma unroll
    for (int r = 0; r < 32; r += 8) {
      const int row = r + srow;
      const unsigned gk = (unsigned)min(k0 + row, len - 1);
      const float4 kv = *(const float4*)(kbase + (gk * (unsigned)(HH * DD)  + scol4));
      const float4 vv = *(const float4*)(vbase + (gk * (unsigned)(HH * DVV) + scol4));
      Ks[row][scol4 + 0] = (__bf16)kv.x;
      Ks[row][scol4 + 1] = (__bf16)kv.y;
      Ks[row][scol4 + 2] = (__bf16)kv.z;
      Ks[row][scol4 + 3] = (__bf16)kv.w;
      Vt[scol4 + 0][row] = (__bf16)vv.x;
      Vt[scol4 + 1][row] = (__bf16)vv.y;
      Vt[scol4 + 2][row] = (__bf16)vv.z;
      Vt[scol4 + 3][row] = (__bf16)vv.w;
    }
    if (k0 + 32 < kend) {
      __builtin_prefetch(kbase + (unsigned)(k0 + 32) * (unsigned)(HH * DD),  0, 1);
      __builtin_prefetch(vbase + (unsigned)(k0 + 32) * (unsigned)(HH * DVV), 0, 1);
    }
    __syncthreads();   // staging visible to all waves

    // wave-uniform per-tile predicates: skip fully-masked chunks
    bool need[2];
    #pragma unroll
    for (int t = 0; t < 2; ++t)
      need[t] = tact[t] && (k0 < q0 + 16 * t + 16 || trz[t]);
    if (!need[0] && !need[1]) continue;

    // ---- S = Q·K^T, both tiles share the bk fragments ----
    #pragma unroll
    for (int s = 0; s < 2; ++s) {
      const v16bf bk0 = *(const v16bf*)&Ks[s * 16 + m][      kh * 16];
      const v16bf bk1 = *(const v16bf*)&Ks[s * 16 + m][32 +  kh * 16];
      const int col    = k0 + s * 16 + m;
      const bool colok = (col < len);
      int ac = col - nctx + 1;
      ac = ac > 0 ? ac : 0;
      ac = ac < max_ids ? ac : max_ids;
      const bool cfree = (ac < max_ids);
      #pragma unroll
      for (int t = 0; t < 2; ++t) {
        if (!need[t]) continue;
        v8f c = {};
        c = __builtin_amdgcn_wmma_f32_16x16x32_bf16(false, aq[t][0], false, bk0,
                                                    (short)0, c, false, false);
        c = __builtin_amdgcn_wmma_f32_16x16x32_bf16(false, aq[t][1], false, bk1,
                                                    (short)0, c, false, false);
        const int q0t = q0 + 16 * t;
        #pragma unroll
        for (int j = 0; j < 8; ++j) {
          const int rowq = q0t + j + 8 * kh;
          // bitwise (non-short-circuit) mask: lane-mask s_and/s_or + one cndmask
          const bool ok = (colok & rvalid[t][j]) &
                          ((rowq == col) | (adjr[t][j] > ac) |
                           (rzero[t][j] & cfree));
          const float x = c[j];
          const float sig = __builtin_amdgcn_rcpf(1.f + __expf(-x));
          const float p = (ok ? rN : 0.f) * x * sig;
          Ss[wave][16 * t + j + 8 * kh][s * 16 + m] = p;
        }
      }
    }

    // ---- re-layout P (C-layout -> A-fragment) per tile ----
    v16bf ap[2];
    #pragma unroll
    for (int t = 0; t < 2; ++t) {
      if (!need[t]) continue;
      const v8f plo = *(const v8f*)&Ss[wave][16 * t + m][      kh * 16];
      const v8f phi = *(const v8f*)&Ss[wave][16 * t + m][8 +  kh * 16];
      v8bf blo, bhi;
      #pragma unroll
      for (int e = 0; e < 8; ++e) {
        blo[e] = (__bf16)plo[e];
        bhi[e] = (__bf16)phi[e];
      }
      ap[t] = cat8(blo, bhi);
    }

    // ---- O += P·V over DV=64, both tiles share the bv fragments ----
    #pragma unroll
    for (int s4 = 0; s4 < 4; ++s4) {
      const v16bf bv = *(const v16bf*)&Vt[s4 * 16 + m][kh * 16];
      #pragma unroll
      for (int t = 0; t < 2; ++t) {
        if (!need[t]) continue;
        acc[t][s4] = __builtin_amdgcn_wmma_f32_16x16x32_bf16(
            false, ap[t], false, bv, (short)0, acc[t][s4], false, false);
      }
    }
  }

  // ---- store O (f32), masked to valid rows ----
  #pragma unroll
  for (int t = 0; t < 2; ++t) {
    if (!tact[t]) continue;
    #pragma unroll
    for (int j = 0; j < 8; ++j) {
      const int rowq = q0 + 16 * t + j + 8 * kh;
      if (rowq < len) {
        float* o = obase + (unsigned)rowq * (unsigned)(HH * DVV);
        o[ 0 + m] = acc[t][0][j];
        o[16 + m] = acc[t][1][j];
        o[32 + m] = acc[t][2][j];
        o[48 + m] = acc[t][3][j];
      }
    }
  }
}

extern "C" void kernel_launch(void* const* d_in, const int* in_sizes, int n_in,
                              void* d_out, int out_size, void* d_ws, size_t ws_size,
                              hipStream_t stream) {
  const float* tq       = (const float*)d_in[0];
  const float* tk       = (const float*)d_in[1];
  const float* tv       = (const float*)d_in[2];
  const int*   offsets  = (const int*)d_in[3];
  const int*   max_seq  = (const int*)d_in[4];
  const int*   num_cand = (const int*)d_in[5];
  const int*   num_ctx  = (const int*)d_in[6];
  float*       out      = (float*)d_out;

  const int B = in_sizes[3] - 1;               // offsets has B+1 entries
  dim3 grid(B * HH * QBLKS);                   // qb bits[2:0], h bits[5:3], b above
  hstu_attn_kernel<<<grid, 128, 0, stream>>>(tq, tk, tv, offsets, max_seq,
                                             num_cand, num_ctx, out);
}
